// Attention_4664334483934
// MI455X (gfx1250) — matrix operand order repaired and checked by reading, not compiled, not run on hardware
//
#include <hip/hip_runtime.h>

// ---------------------------------------------------------------------------
// MI455X (gfx1250) attention: bf16 WMMA (v_wmma_f32_16x16x32_bf16) everywhere,
// async global->LDS (ASYNCcnt) double-buffered K/V staging in the attention
// kernel, register-pipelined GEMMs.
// B=8, C=512, N=1024, heads=8, d=64.  Wave32.
// ---------------------------------------------------------------------------

typedef unsigned short u16t;
typedef __attribute__((ext_vector_type(16))) __bf16 v16bf;
typedef __attribute__((ext_vector_type(8)))  float  v8f;
typedef __attribute__((ext_vector_type(4)))  unsigned int v4u;

union Frag { v16bf v; v4u q[2]; };

#define BATCH   8
#define CDIM    512
#define NSEQ    1024
#define HEADS   8
#define DHEAD   64
#define INNER   512           // HEADS*DHEAD
#define FUSED   1536          // q(512) + k(512) + v(512) output cols
#define SCALE   0.125f        // 64^-0.5

__device__ inline u16t f2bf(float x) {
    union { float f; unsigned u; } v; v.f = x;
    unsigned r = v.u + 0x7FFFu + ((v.u >> 16) & 1u);   // round-to-nearest-even
    return (u16t)(r >> 16);
}

// A-fragment (16x32 bf16, M=lane&15): 16 contiguous K halves at hi*16
__device__ inline v16bf loadA(const u16t* rowk, int hi) {
    Frag f;
    const v4u* p = (const v4u*)(rowk + hi * 16);
    f.q[0] = p[0];
    f.q[1] = p[1];
    return f.v;
}

// B-fragment (32x16 bf16, N=lane&15): 16 contiguous K halves at hi*16
__device__ inline v16bf loadB(const u16t* colk, int hi) {
    Frag f;
    const v4u* p = (const v4u*)(colk + hi * 16);
    f.q[0] = p[0];
    f.q[1] = p[1];
    return f.v;
}

__device__ inline v8f wmma_bf16(v16bf a, v16bf b, v8f c) {
    return __builtin_amdgcn_wmma_f32_16x16x32_bf16(
        false, a, false, b, (short)0, c, false, false);
}

// async global -> LDS, 16B per lane, tracked by ASYNCcnt (gfx1250)
__device__ inline void async_b128(unsigned lds_off, unsigned long long gaddr) {
    asm volatile("global_load_async_to_lds_b128 %0, %1, off"
                 :: "v"(lds_off), "v"(gaddr) : "memory");
}

// ---------------------------------------------------------------------------
// 1) x (B,C,N) f32  ->  xt (B,N,C) bf16   (LDS-tiled transpose)
// ---------------------------------------------------------------------------
__global__ __launch_bounds__(256) void k_convert_x(const float* __restrict__ x,
                                                   u16t* __restrict__ xt) {
    __shared__ float tile[32][33];
    const int b  = blockIdx.z;
    const int c0 = blockIdx.y * 32;
    const int n0 = blockIdx.x * 32;
    for (int i = threadIdx.y; i < 32; i += 8)
        tile[i][threadIdx.x] = x[((size_t)b * CDIM + c0 + i) * NSEQ + n0 + threadIdx.x];
    __syncthreads();
    for (int i = threadIdx.y; i < 32; i += 8)
        xt[((size_t)b * NSEQ + n0 + i) * CDIM + c0 + threadIdx.x] = f2bf(tile[threadIdx.x][i]);
}

// ---------------------------------------------------------------------------
// 2) weights -> transposed bf16: WfT (1536,512) = [Wq | Wkv]^T, WoT (512,512)
// ---------------------------------------------------------------------------
__global__ __launch_bounds__(256) void k_convert_w(const float* __restrict__ Wq,
                                                   const float* __restrict__ Wkv,
                                                   const float* __restrict__ Wo,
                                                   u16t* __restrict__ WfT,
                                                   u16t* __restrict__ WoT) {
    const int idx = blockIdx.x * 256 + threadIdx.x;
    const int t1  = FUSED * CDIM;                 // 786432
    if (idx < t1) {
        const int o = idx / CDIM, c = idx % CDIM;
        const float v = (o < INNER) ? Wq[(size_t)c * INNER + o]
                                    : Wkv[(size_t)c * (2 * INNER) + (o - INNER)];
        WfT[idx] = f2bf(v);
    } else {
        const int j = idx - t1;
        const int o = j / CDIM, c = j % CDIM;
        WoT[j] = f2bf(Wo[(size_t)c * CDIM + o]);
    }
}

// ---------------------------------------------------------------------------
// 3) fused QKV GEMM: Y = xt @ [Wq|Wkv] -> q (b,h,n,d)*SCALE, k (b,h,n,d),
//    vT (b,h,d,n), all bf16.  One wave = 16(M) x 64(N), K-step 32, register
//    ping-pong so WMMAs overlap the next step's global loads.
// ---------------------------------------------------------------------------
__global__ __launch_bounds__(256) void k_qkv_gemm(const u16t* __restrict__ xt,
                                                  const u16t* __restrict__ WfT,
                                                  u16t* __restrict__ qb,
                                                  u16t* __restrict__ kb,
                                                  u16t* __restrict__ vTb) {
    const int lane = threadIdx.x & 31, wave = threadIdx.x >> 5;
    const int gw   = blockIdx.x * 8 + wave;          // 12288 waves
    const int nt   = gw % 24;                        // 64-col group in [0,1536)
    const int mt   = gw / 24;                        // 16-row tile in [0,8192)
    const int row  = lane & 15, hi = (lane >> 4) & 1;

    const u16t* arow = xt + ((size_t)mt * 16 + row) * CDIM;
    const u16t* bbase[4];
#pragma unroll
    for (int t = 0; t < 4; ++t)
        bbase[t] = WfT + (size_t)(nt * 64 + t * 16 + row) * CDIM;

    v8f acc[4] = {};
    v16bf aN = loadA(arow, hi);
    v16bf bN[4];
#pragma unroll
    for (int t = 0; t < 4; ++t) bN[t] = loadB(bbase[t], hi);

    for (int c0 = 0; c0 < CDIM; c0 += 32) {
        const v16bf aC = aN;
        const v16bf b0 = bN[0], b1 = bN[1], b2 = bN[2], b3 = bN[3];
        const int cn = c0 + 32;
        if (cn < CDIM) {
            __builtin_prefetch(arow + cn + 32);
            aN = loadA(arow + cn, hi);
#pragma unroll
            for (int t = 0; t < 4; ++t) bN[t] = loadB(bbase[t] + cn, hi);
        }
        acc[0] = wmma_bf16(aC, b0, acc[0]);
        acc[1] = wmma_bf16(aC, b1, acc[1]);
        acc[2] = wmma_bf16(aC, b2, acc[2]);
        acc[3] = wmma_bf16(aC, b3, acc[3]);
    }
#pragma unroll
    for (int t = 0; t < 4; ++t) {
        const int o = nt * 64 + t * 16 + row;
#pragma unroll
        for (int r = 0; r < 8; ++r) {
            const int gr = mt * 16 + r + hi * 8;
            const int b  = gr >> 10, n = gr & 1023;
            const float val = acc[t][r];
            if (o < INNER) {                       // Q (scaled)
                const int h = o >> 6, d = o & 63;
                qb[(((size_t)b * HEADS + h) * NSEQ + n) * DHEAD + d] = f2bf(val * SCALE);
            } else if (o < 2 * INNER) {            // K
                const int oo = o - INNER, h = oo >> 6, d = oo & 63;
                kb[(((size_t)b * HEADS + h) * NSEQ + n) * DHEAD + d] = f2bf(val);
            } else {                               // V transposed (d, n)
                const int oo = o - 2 * INNER, h = oo >> 6, d = oo & 63;
                vTb[(((size_t)b * HEADS + h) * DHEAD + d) * NSEQ + n] = f2bf(val);
            }
        }
    }
}

// ---------------------------------------------------------------------------
// 4) flash attention: block = 8 waves, all on the same (b,h), one 16-query
//    tile each.  K/V j-tiles are staged into block-shared LDS with async
//    global->LDS b128 copies (double buffered, ASYNCcnt-pipelined).
// ---------------------------------------------------------------------------
__global__ __launch_bounds__(256) void k_attn(const u16t* __restrict__ qb,
                                              const u16t* __restrict__ kb,
                                              const u16t* __restrict__ vTb,
                                              u16t* __restrict__ aout) {
    __shared__ __align__(128) u16t Kt[2][32 * 64];   // 2 x 4KB, k rows (j, d)
    __shared__ __align__(128) u16t Vt[2][64 * 32];   // 2 x 4KB, vT rows (d, j)
    __shared__ __align__(64)  u16t Plds[8][16][32];  // per-wave P tile

    const int lane = threadIdx.x & 31, wave = threadIdx.x >> 5;
    const int gw   = blockIdx.x * 8 + wave;          // 4096 waves
    const int mt   = gw & 63;                        // query tile
    const int bh   = gw >> 6;                        // same for whole block
    const int row  = lane & 15, hi = (lane >> 4) & 1;

    const u16t* qp = qb  + ((size_t)bh * NSEQ + mt * 16) * DHEAD;
    const u16t* kp = kb  + (size_t)bh * NSEQ * DHEAD;
    const u16t* vp = vTb + (size_t)bh * DHEAD * NSEQ;

    const unsigned ldsK[2] = { (unsigned)(size_t)&Kt[0][0], (unsigned)(size_t)&Kt[1][0] };
    const unsigned ldsV[2] = { (unsigned)(size_t)&Vt[0][0], (unsigned)(size_t)&Vt[1][0] };

    // q fragments held in registers for the whole pass
    const u16t* qrow = qp + (size_t)row * DHEAD;
    const v16bf qa0 = loadA(qrow, hi);               // d = 0..31
    const v16bf qa1 = loadA(qrow + 32, hi);          // d = 32..63

    float m[8], l[8];
#pragma unroll
    for (int r = 0; r < 8; ++r) { m[r] = -3.0e38f; l[r] = 0.0f; }
    v8f acc[4] = {};

    // stage helper (wave 0 only): 8+8 async b128 issues = 16 per tile pair
    auto stage = [&](int buf, int j0) {
        // K tile: 4KB contiguous at kp + j0*64 halves
        {
            unsigned long long g = (unsigned long long)(kp + (size_t)j0 * DHEAD) + (unsigned)(lane * 16);
            unsigned l0 = ldsK[buf] + (unsigned)(lane * 16);
#pragma unroll
            for (int i = 0; i < 8; ++i) async_b128(l0 + i * 512, g + i * 512);
        }
        // V tile: 64 rows x 64B, global row stride 2048B
        {
            const int r8 = lane >> 2, cq = lane & 3;
            unsigned long long g = (unsigned long long)(vp + (size_t)r8 * NSEQ + j0) + (unsigned)(cq * 16);
            unsigned l0 = ldsV[buf] + (unsigned)(r8 * 64 + cq * 16);
#pragma unroll
            for (int i = 0; i < 8; ++i) async_b128(l0 + i * 512, g + i * 16384);
        }
    };

    if (wave == 0) stage(0, 0);

    for (int j0 = 0, idx = 0; j0 < NSEQ; j0 += 32, ++idx) {
        const int cur = idx & 1, nxt = cur ^ 1;
        const bool has_next = (j0 + 32) < NSEQ;
        if (wave == 0 && has_next) stage(nxt, j0 + 32);
        if (has_next) asm volatile("s_wait_asynccnt 0x10" ::: "memory");
        else          asm volatile("s_wait_asynccnt 0x0"  ::: "memory");
        __syncthreads();                               // cur tile visible

        const u16t* Kc = &Kt[cur][0];
        const u16t* Vc = &Vt[cur][0];

        // S tiles: 16 queries x 32 keys (B-frags from LDS)
        v8f s0 = {}, s1 = {};
        s0 = wmma_bf16(qa0, loadB(Kc + (size_t)row * 64, hi), s0);
        s0 = wmma_bf16(qa1, loadB(Kc + (size_t)row * 64 + 32, hi), s0);
        s1 = wmma_bf16(qa0, loadB(Kc + (size_t)(16 + row) * 64, hi), s1);
        s1 = wmma_bf16(qa1, loadB(Kc + (size_t)(16 + row) * 64 + 32, hi), s1);

        // online softmax (row i = r + hi*8 lives in one 16-lane group)
#pragma unroll
        for (int r = 0; r < 8; ++r) {
            const float a = s0[r], b = s1[r];
            float mx = fmaxf(a, b);
            mx = fmaxf(mx, __shfl_xor(mx, 1));
            mx = fmaxf(mx, __shfl_xor(mx, 2));
            mx = fmaxf(mx, __shfl_xor(mx, 4));
            mx = fmaxf(mx, __shfl_xor(mx, 8));
            const float mnew  = fmaxf(m[r], mx);
            const float p0    = __expf(a - mnew);
            const float p1    = __expf(b - mnew);
            const float alpha = __expf(m[r] - mnew);
            m[r] = mnew;
            float rs = p0 + p1;
            rs += __shfl_xor(rs, 1);
            rs += __shfl_xor(rs, 2);
            rs += __shfl_xor(rs, 4);
            rs += __shfl_xor(rs, 8);
            l[r] = l[r] * alpha + rs;
#pragma unroll
            for (int t = 0; t < 4; ++t) acc[t][r] *= alpha;
            const int i = r + hi * 8;
            Plds[wave][i][row]      = f2bf(p0);
            Plds[wave][i][16 + row] = f2bf(p1);
        }
        asm volatile("s_wait_dscnt 0x0" ::: "memory"); // P stores -> P loads
        const v16bf pa = loadA(&Plds[wave][row][0], hi);
#pragma unroll
        for (int t = 0; t < 4; ++t)
            acc[t] = wmma_bf16(pa, loadB(Vc + (size_t)(t * 16 + row) * 32, hi), acc[t]);

        __syncthreads();                               // done reading cur
    }

    // normalize + scatter to attn_out (b, n, h*d) bf16
    const int b = bh >> 3, h = bh & 7;
#pragma unroll
    for (int r = 0; r < 8; ++r) {
        const int i = r + hi * 8;
        const int n = mt * 16 + i;
        const float inv = 1.0f / l[r];
        u16t* op = aout + ((size_t)(b * NSEQ + n)) * INNER + h * DHEAD;
#pragma unroll
        for (int t = 0; t < 4; ++t) op[t * 16 + row] = f2bf(acc[t][r] * inv);
    }
}

// ---------------------------------------------------------------------------
// 5) output projection: out = attn_out @ Wo + bo   (f32 result)
// ---------------------------------------------------------------------------
__global__ __launch_bounds__(256) void k_out_gemm(const u16t* __restrict__ aout,
                                                  const u16t* __restrict__ WoT,
                                                  const float* __restrict__ bo,
                                                  float* __restrict__ out) {
    const int lane = threadIdx.x & 31, wave = threadIdx.x >> 5;
    const int gw   = blockIdx.x * 8 + wave;          // 4096 waves
    const int nt   = gw & 7;                         // 64-col group in [0,512)
    const int mt   = gw >> 3;                        // 16-row tile in [0,8192)
    const int row  = lane & 15, hi = (lane >> 4) & 1;

    const u16t* arow = aout + ((size_t)mt * 16 + row) * INNER;
    const u16t* bbase[4];
#pragma unroll
    for (int t = 0; t < 4; ++t)
        bbase[t] = WoT + (size_t)(nt * 64 + t * 16 + row) * INNER;

    v8f acc[4] = {};
    v16bf aN = loadA(arow, hi);
    v16bf bN[4];
#pragma unroll
    for (int t = 0; t < 4; ++t) bN[t] = loadB(bbase[t], hi);

    for (int c0 = 0; c0 < INNER; c0 += 32) {
        const v16bf aC = aN;
        const v16bf b0 = bN[0], b1 = bN[1], b2 = bN[2], b3 = bN[3];
        const int cn = c0 + 32;
        if (cn < INNER) {
            __builtin_prefetch(arow + cn + 32);
            aN = loadA(arow + cn, hi);
#pragma unroll
            for (int t = 0; t < 4; ++t) bN[t] = loadB(bbase[t] + cn, hi);
        }
        acc[0] = wmma_bf16(aC, b0, acc[0]);
        acc[1] = wmma_bf16(aC, b1, acc[1]);
        acc[2] = wmma_bf16(aC, b2, acc[2]);
        acc[3] = wmma_bf16(aC, b3, acc[3]);
    }
#pragma unroll
    for (int t = 0; t < 4; ++t) {
        const int o = nt * 64 + t * 16 + row;
        const float bias = bo[o];
#pragma unroll
        for (int r = 0; r < 8; ++r) {
            const int gr = mt * 16 + r + hi * 8;
            out[(size_t)gr * CDIM + o] = acc[t][r] + bias;
        }
    }
}

// ---------------------------------------------------------------------------
// launch: inputs {x, Wq, Wkv, Wo, bo} f32; output (B,N,C) f32.
// Workspace layout (bytes), ~42 MB total:
//   xt 0..8M | q 8M | k 16M | vT 24M | aout 32M | WfT 40M | WoT 41.5M
// ---------------------------------------------------------------------------
extern "C" void kernel_launch(void* const* d_in, const int* in_sizes, int n_in,
                              void* d_out, int out_size, void* d_ws, size_t ws_size,
                              hipStream_t stream) {
    (void)in_sizes; (void)n_in; (void)out_size; (void)ws_size;
    const float* x   = (const float*)d_in[0];
    const float* Wq  = (const float*)d_in[1];
    const float* Wkv = (const float*)d_in[2];
    const float* Wo  = (const float*)d_in[3];
    const float* bo  = (const float*)d_in[4];
    float* out = (float*)d_out;

    char* ws = (char*)d_ws;
    u16t* xt   = (u16t*)(ws + 0);
    u16t* qb   = (u16t*)(ws + 8388608);
    u16t* kb   = (u16t*)(ws + 16777216);
    u16t* vTb  = (u16t*)(ws + 25165824);
    u16t* aout = (u16t*)(ws + 33554432);
    u16t* WfT  = (u16t*)(ws + 41943040);
    u16t* WoT  = (u16t*)(ws + 43515904);

    k_convert_x<<<dim3(32, 16, 8), dim3(32, 8), 0, stream>>>(x, xt);
    k_convert_w<<<4096, 256, 0, stream>>>(Wq, Wkv, Wo, WfT, WoT);
    k_qkv_gemm<<<1536, 256, 0, stream>>>(xt, WfT, qb, kb, vTb);
    k_attn<<<512, 256, 0, stream>>>(qb, kb, vTb, aout);
    k_out_gemm<<<512, 256, 0, stream>>>(aout, WoT, bo, out);
}
